// MeshNet_11218454577411
// MI455X (gfx1250) — matrix operand order repaired and checked by reading, not compiled, not run on hardware
//
#include <hip/hip_runtime.h>

// ---------------------------------------------------------------------------
// MeshNet on MI455X (gfx1250, wave32, WMMA 16x16x32 f16->f32)
//
// GEMM: one wave computes a 32x32 output block (2x2 register tiles, 4
// accumulators). Per k-step: 8 unconditional global_load_b128 pairs feed 4
// v_wmma_f32_16x16x32_f16 (2x fragment reuse vs 16x16/wave).
//   A  [Mpad x Kpad] row-major, zero-padded (Mpad mult 32, Kpad mult 32)
//   B^T[Npad x Kpad] row-major, zero-padded (Npad mult 32)
// Needs ~350 MB workspace.
// ---------------------------------------------------------------------------

typedef __attribute__((ext_vector_type(16))) _Float16 v16h;
typedef __attribute__((ext_vector_type(8)))  _Float16 v8h;
typedef __attribute__((ext_vector_type(8)))  float    v8f;
typedef _Float16 half_t;

#define NV    6890
#define NVP   6912       // padded to 32 (A-matrix rows)
#define NE    41328
#define NB    32
#define NC    720
#define NCP   736        // padded to 32
#define NL    56
#define NN1   1792       // NB*NL
#define NZ    12544      // 56*224 (already mult of 32)
#define NP    224
#define NXYZ  192        // 168 padded to 32-multiple

// ---------------------------------------------------------------------------
// 1) Fused mean reductions over img_feat.  grid = NB*NC blocks, 64 threads.
//    mxT[(b*56+w)*736 + c] = mean_h img[b,c,h,w]    (B^T for heatmap_x GEMM)
//    myT[(b*56+h)*736 + c] = mean_w img[b,c,h,w]    (B^T for heatmap_y GEMM)
//    gT [b*736 + c]        = mean_hw img[b,c,:,:]   (B^T for z/w head GEMM1)
//    (pad columns 720..735 pre-zeroed by memset)
// ---------------------------------------------------------------------------
__global__ __launch_bounds__(64) void reduce_means(
    const float* __restrict__ img, half_t* __restrict__ mxT,
    half_t* __restrict__ myT, half_t* __restrict__ gT)
{
  int bc = blockIdx.x;                // b*720 + c
  int b = bc / NC, c = bc % NC;
  const float* p = img + (size_t)bc * (NL * NL);
  int t = threadIdx.x;
  __shared__ float s[64];
  float cs = 0.f;
  if (t < NL) {
    for (int h = 0; h < NL; ++h) cs += p[h * NL + t];     // coalesced across t
    mxT[(size_t)(b * NL + t) * NCP + c] = (half_t)(cs * (1.f / 56.f));
  }
  s[t] = (t < NL) ? cs : 0.f;
  __syncthreads();
  if (t < NL) {
    float rs = 0.f;
    for (int w = 0; w < NL; ++w) rs += p[t * NL + w];     // tile is cache-hot
    myT[(size_t)(b * NL + t) * NCP + c] = (half_t)(rs * (1.f / 56.f));
  }
  if (t == 0) {
    float tot = 0.f;
    for (int i = 0; i < NL; ++i) tot += s[i];
    gT[(size_t)b * NCP + c] = (half_t)(tot * (1.f / 3136.f));
  }
}

// ---------------------------------------------------------------------------
// Weight conversions with zero padding
// ---------------------------------------------------------------------------
// A-style: dst[dr x dc] = pad(src[sr x sc])
__global__ void cvt_pad2d(const float* __restrict__ src, half_t* __restrict__ dst,
                          int sr, int sc, int dr, int dc)
{
  int i = blockIdx.x * blockDim.x + threadIdx.x;
  if (i >= dr * dc) return;
  int r = i / dc, c = i % dc;
  dst[i] = (r < sr && c < sc) ? (half_t)src[r * sc + c] : (half_t)0.f;
}

// B^T-style: dst[n*kp + k] = pad(src[k*N + n]) ; dst is [nd x kp]
__global__ void cvt_bT(const float* __restrict__ src, half_t* __restrict__ dst,
                       int K, int N, int kp, int nd)
{
  int i = blockIdx.x * blockDim.x + threadIdx.x;
  if (i >= nd * kp) return;
  int n = i / kp, k = i % kp;
  dst[i] = (k < K && n < N) ? (half_t)src[k * N + n] : (half_t)0.f;
}

// ---------------------------------------------------------------------------
// Epilogue store helper (epi is wave-uniform):
//   0: xyz f16:  xyz[((col/56)*NV + row)*192 + off + col%56] = v+bias[row]
//   1: zBt f16:  dst[(col*56 + row%56)*224 + row/56]         = v+bias[row]
//   2: hw  f32:  hw[((col/56)*NV + row)*56 + col%56]         = v+bias[row]
//   3: f32:      outF[row*ldo + col]                         = v
// ---------------------------------------------------------------------------
__device__ __forceinline__ void epi_store(
    float v, int row, int col, int M,
    const float* __restrict__ bias,
    float* __restrict__ outF, half_t* __restrict__ outH,
    int epi, int off, int ldo)
{
  if (row >= M) return;
  if (epi == 0) {
    v += bias[row];
    outH[((size_t)(col / 56) * NV + row) * NXYZ + off + (col % 56)] = (half_t)v;
  } else if (epi == 1) {
    v += bias[row];
    outH[((size_t)col * 56 + (row % 56)) * NP + (row / 56)] = (half_t)v;
  } else if (epi == 2) {
    v += bias[row];
    outF[((size_t)(col / 56) * NV + row) * 56 + (col % 56)] = v;
  } else {
    outF[(size_t)row * ldo + col] = v;
  }
}

// ---------------------------------------------------------------------------
// WMMA GEMM, 2x2 register blocking: one wave = 32x32 output block.
// A [Mpad x lda] row-major (Mpad mult 32), Bt [Npad x ldb] row-major
// (= B column-major, Npad mult 32), Kp loop bound mult of 32, zero padding.
// ---------------------------------------------------------------------------
__global__ __launch_bounds__(256) void wmma_gemm_f16(
    const half_t* __restrict__ A, int lda,
    const half_t* __restrict__ Bt, int ldb,
    int M, int N, int Kp,
    const float* __restrict__ bias,
    float* __restrict__ outF, half_t* __restrict__ outH,
    int epi, int off, int ldo)
{
  const int Nt = N >> 5;                            // 32-wide blocks
  const int Mt = (M + 31) >> 5;
  const int wid = blockIdx.x * (blockDim.x >> 5) + (threadIdx.x >> 5);
  if (wid >= Mt * Nt) return;                       // wave-uniform
  const int mt = wid / Nt, nt = wid % Nt;
  const int lane  = threadIdx.x & 31;
  const int lrow  = lane & 15;
  const int khalf = lane >> 4;                      // 0 or 1
  const int m0 = mt << 5, n0 = nt << 5;

  // A fragment rows m0+lrow (+16), K runs {khalf*8+0..7, 16+khalf*8+0..7}
  const half_t* Ap0 = A  + (size_t)(m0 + lrow) * lda + khalf * 8;
  const half_t* Ap1 = Ap0 + (size_t)16 * lda;
  // B fragment cols n0+lrow (+16), K runs {khalf*8+0..7, 16+khalf*8+0..7}
  const half_t* Bp0 = Bt + (size_t)(n0 + lrow) * ldb + khalf * 8;
  const half_t* Bp1 = Bp0 + (size_t)16 * ldb;

  v8f acc00 = {}, acc01 = {}, acc10 = {}, acc11 = {};
  for (int k0 = 0; k0 < Kp; k0 += 32) {
    v8h a0l = *(const v8h*)(Ap0);
    v8h a0h = *(const v8h*)(Ap0 + 16);
    v8h a1l = *(const v8h*)(Ap1);
    v8h a1h = *(const v8h*)(Ap1 + 16);
    v8h b0l = *(const v8h*)(Bp0);
    v8h b0h = *(const v8h*)(Bp0 + 16);
    v8h b1l = *(const v8h*)(Bp1);
    v8h b1h = *(const v8h*)(Bp1 + 16);
    v16h a0 = __builtin_shufflevector(a0l, a0h, 0,1,2,3,4,5,6,7,8,9,10,11,12,13,14,15);
    v16h a1 = __builtin_shufflevector(a1l, a1h, 0,1,2,3,4,5,6,7,8,9,10,11,12,13,14,15);
    v16h b0 = __builtin_shufflevector(b0l, b0h, 0,1,2,3,4,5,6,7,8,9,10,11,12,13,14,15);
    v16h b1 = __builtin_shufflevector(b1l, b1h, 0,1,2,3,4,5,6,7,8,9,10,11,12,13,14,15);
    acc00 = __builtin_amdgcn_wmma_f32_16x16x32_f16(false, a0, false, b0, (short)0, acc00, false, false);
    acc01 = __builtin_amdgcn_wmma_f32_16x16x32_f16(false, a0, false, b1, (short)0, acc01, false, false);
    acc10 = __builtin_amdgcn_wmma_f32_16x16x32_f16(false, a1, false, b0, (short)0, acc10, false, false);
    acc11 = __builtin_amdgcn_wmma_f32_16x16x32_f16(false, a1, false, b1, (short)0, acc11, false, false);
    Ap0 += 32; Ap1 += 32; Bp0 += 32; Bp1 += 32;
  }

  // Epilogue: VGPR r -> row m0+r+8*khalf (+16), col n0+lrow (+16)
  for (int r = 0; r < 8; ++r) {
    const int row0 = m0 + r + (khalf << 3);
    const int col0 = n0 + lrow;
    epi_store(acc00[r], row0,      col0,      M, bias, outF, outH, epi, off, ldo);
    epi_store(acc01[r], row0,      col0 + 16, M, bias, outF, outH, epi, off, ldo);
    epi_store(acc10[r], row0 + 16, col0,      M, bias, outF, outH, epi, off, ldo);
    epi_store(acc11[r], row0 + 16, col0 + 16, M, bias, outF, outH, epi, off, ldo);
  }
}

// ---------------------------------------------------------------------------
// GCN normalization (PyG gcn_norm with self-loops, edge_weight = 1)
// ---------------------------------------------------------------------------
__global__ void deg_init(float* __restrict__ deg, int n)
{
  int i = blockIdx.x * blockDim.x + threadIdx.x;
  if (i < n) deg[i] = 1.f;                           // self loop
}
__global__ void deg_accum(const int* __restrict__ dst, float* __restrict__ deg, int e)
{
  int i = blockIdx.x * blockDim.x + threadIdx.x;
  if (i < e) atomicAdd(&deg[dst[i]], 1.f);
}
__global__ void deg_rsqrt(float* __restrict__ deg, int n)
{
  int i = blockIdx.x * blockDim.x + threadIdx.x;
  if (i < n) deg[i] = rsqrtf(deg[i]);                // deg >= 1 always
}
__global__ void norm_kernel(const int* __restrict__ src, const int* __restrict__ dst,
                            const float* __restrict__ dinv, float* __restrict__ nrm,
                            int e, int v)
{
  int i = blockIdx.x * blockDim.x + threadIdx.x;
  if (i >= e + v) return;
  int s = (i < e) ? src[i] : (i - e);
  int d = (i < e) ? dst[i] : (i - e);
  nrm[i] = dinv[s] * dinv[d];
}

// One block per (edge or self-loop); loops over batch x channels.
__global__ __launch_bounds__(256) void gcn_scatter(
    const int* __restrict__ src, const int* __restrict__ dst,
    const float* __restrict__ nrm,
    const float* __restrict__ xw, float* __restrict__ agg,
    int e, int nch, int ld)
{
  const int i = blockIdx.x;
  const int s = (i < e) ? src[i] : (i - e);
  const int d = (i < e) ? dst[i] : (i - e);
  const float w = nrm[i];
  const int tot = NB * nch;
  for (int t = threadIdx.x; t < tot; t += blockDim.x) {
    const int b = t / nch, ch = t % nch;
    const float v = xw[((size_t)b * NV + s) * ld + ch] * w;
    atomicAdd(&agg[((size_t)b * NV + d) * ld + ch], v);
  }
}

__global__ void bias_lrelu(const float* __restrict__ agg, const float* __restrict__ bias,
                           half_t* __restrict__ h, int rows, int nch)
{
  int i = blockIdx.x * blockDim.x + threadIdx.x;
  if (i >= rows * nch) return;
  int c = i % nch;
  float v = agg[i] + bias[c];
  v = (v > 0.f) ? v : 0.01f * v;
  h[i] = (half_t)v;
}

// out[r,0:3] = agg3[r*32 + 0:3] + bg3 ; out[r,3] = sigmoid(mean_l hw[r*56+l])
__global__ void finalize_kernel(const float* __restrict__ agg3,
                                const float* __restrict__ bg3,
                                const float* __restrict__ hw,
                                float* __restrict__ out, int total)
{
  int r = blockIdx.x * blockDim.x + threadIdx.x;
  if (r >= total) return;
  const float* p = hw + (size_t)r * 56;
  float s = 0.f;
  for (int l = 0; l < 56; ++l) s += p[l];
  s *= (1.f / 56.f);
  out[(size_t)r * 4 + 0] = agg3[(size_t)r * 32 + 0] + bg3[0];
  out[(size_t)r * 4 + 1] = agg3[(size_t)r * 32 + 1] + bg3[1];
  out[(size_t)r * 4 + 2] = agg3[(size_t)r * 32 + 2] + bg3[2];
  out[(size_t)r * 4 + 3] = 1.f / (1.f + __expf(-s));
}

// ---------------------------------------------------------------------------
// Host orchestration
// ---------------------------------------------------------------------------
extern "C" void kernel_launch(void* const* d_in, const int* in_sizes, int n_in,
                              void* d_out, int out_size, void* d_ws, size_t ws_size,
                              hipStream_t stream)
{
  const float* img = (const float*)d_in[0];
  const int*   ei  = (const int*)d_in[1];
  const int* e_src = ei;
  const int* e_dst = ei + NE;
  const float* Wx  = (const float*)d_in[2];  const float* bx  = (const float*)d_in[3];
  const float* Wy  = (const float*)d_in[4];  const float* by  = (const float*)d_in[5];
  const float* Wz1 = (const float*)d_in[6];  const float* bz1 = (const float*)d_in[7];
  const float* Wz2 = (const float*)d_in[8];  const float* bz2 = (const float*)d_in[9];
  const float* Ww1 = (const float*)d_in[10]; const float* bw1 = (const float*)d_in[11];
  const float* Ww2 = (const float*)d_in[12]; const float* bw2 = (const float*)d_in[13];
  const float* Wg1 = (const float*)d_in[14]; const float* bg1 = (const float*)d_in[15];
  const float* Wg2 = (const float*)d_in[16]; const float* bg2 = (const float*)d_in[17];
  const float* Wg3 = (const float*)d_in[18]; const float* bg3 = (const float*)d_in[19];
  float* out = (float*)d_out;

  // --- workspace carve-up (256B aligned) ---
  char* ws = (char*)d_ws;
  size_t off = 0;
  auto alloc = [&](size_t bytes) -> void* {
    void* p = ws + off;
    off = (off + bytes + 255) & ~(size_t)255;
    return p;
  };
  half_t* mxT   = (half_t*)alloc((size_t)NN1 * NCP * 2);      // B^T heatmap_x
  half_t* myT   = (half_t*)alloc((size_t)NN1 * NCP * 2);      // B^T heatmap_y
  half_t* gT    = (half_t*)alloc((size_t)NB * NCP * 2);       // B^T z/w head 1
  half_t* Wx16  = (half_t*)alloc((size_t)NVP * NCP * 2);
  half_t* Wy16  = (half_t*)alloc((size_t)NVP * NCP * 2);
  half_t* Wz116 = (half_t*)alloc((size_t)NZ * NCP * 2);
  half_t* Ww116 = (half_t*)alloc((size_t)NZ * NCP * 2);
  half_t* Wz216 = (half_t*)alloc((size_t)NVP * NP * 2);
  half_t* Ww216 = (half_t*)alloc((size_t)NVP * NP * 2);
  half_t* Wg1t  = (half_t*)alloc((size_t)64 * 192 * 2);       // B^T [64 x 192]
  half_t* Wg2t  = (half_t*)alloc((size_t)32 * 64 * 2);        // B^T [32 x 64]
  half_t* Wg3t  = (half_t*)alloc((size_t)32 * 32 * 2);        // B^T [32 x 32], N pad 3->32
  half_t* zBt   = (half_t*)alloc((size_t)NN1 * NP * 2);       // B^T heatmap_z2
  half_t* wBt   = (half_t*)alloc((size_t)NN1 * NP * 2);       // B^T heatmap_w2
  half_t* xyz   = (half_t*)alloc((size_t)NB * NV * NXYZ * 2); // A for GCN L1
  float*  hw    = (float*) alloc((size_t)NB * NV * 56 * 4);
  float*  xw    = (float*) alloc((size_t)NB * NV * 64 * 4);   // reused L1/L2/L3
  float*  agg   = (float*) alloc((size_t)NB * NV * 64 * 4);   // reused L1/L2/L3
  half_t* hbuf  = (half_t*)alloc((size_t)NB * NV * 64 * 2);   // h1 then h2
  float*  deg   = (float*) alloc((size_t)NV * 4);
  float*  nrm   = (float*) alloc((size_t)(NE + NV) * 4);
  (void)ws_size; (void)n_in; (void)in_sizes; (void)out_size;

  auto grid1 = [](int n) { return (n + 255) / 256; };
  auto gemm = [&](const half_t* A, int lda, const half_t* Bm, int ldb,
                  int M, int N, int Kp, const float* bias,
                  float* oF, half_t* oH, int epi, int xoff, int ldo) {
    int tiles = ((M + 31) / 32) * (N / 32);           // 32x32 blocks per wave
    int blocks = (tiles + 7) / 8;                     // 8 waves / block
    wmma_gemm_f16<<<blocks, 256, 0, stream>>>(A, lda, Bm, ldb, M, N, Kp,
                                              bias, oF, oH, epi, xoff, ldo);
  };

  // 0) zero the pad regions of activation staging buffers
  hipMemsetAsync(mxT, 0, (size_t)NN1 * NCP * 2, stream);
  hipMemsetAsync(myT, 0, (size_t)NN1 * NCP * 2, stream);
  hipMemsetAsync(gT,  0, (size_t)NB * NCP * 2, stream);
  hipMemsetAsync(xyz, 0, (size_t)NB * NV * NXYZ * 2, stream);

  // 1) means (single pass over 289 MB of img_feat)
  reduce_means<<<NB * NC, 64, 0, stream>>>(img, mxT, myT, gT);

  // 2) weight conversions (A matrices padded; GCN weights transposed)
  cvt_pad2d<<<grid1(NVP * NCP), 256, 0, stream>>>(Wx,  Wx16,  NV, NC, NVP, NCP);
  cvt_pad2d<<<grid1(NVP * NCP), 256, 0, stream>>>(Wy,  Wy16,  NV, NC, NVP, NCP);
  cvt_pad2d<<<grid1(NZ * NCP), 256, 0, stream>>>(Wz1, Wz116, NZ, NC, NZ, NCP);
  cvt_pad2d<<<grid1(NZ * NCP), 256, 0, stream>>>(Ww1, Ww116, NZ, NC, NZ, NCP);
  cvt_pad2d<<<grid1(NVP * NP), 256, 0, stream>>>(Wz2, Wz216, NV, NP, NVP, NP);
  cvt_pad2d<<<grid1(NVP * NP), 256, 0, stream>>>(Ww2, Ww216, NV, NP, NVP, NP);
  cvt_bT<<<grid1(64 * 192), 256, 0, stream>>>(Wg1, Wg1t, 168, 64, 192, 64);
  cvt_bT<<<grid1(32 * 64), 256, 0, stream>>>(Wg2, Wg2t, 64, 32, 64, 32);
  cvt_bT<<<grid1(32 * 32), 256, 0, stream>>>(Wg3, Wg3t, 32, 3, 32, 32);

  // 3) conv1x1 heads as WMMA GEMMs
  gemm(Wx16, NCP, mxT, NCP, NV, NN1, NCP, bx, nullptr, xyz, 0, 0, 0);   // x
  gemm(Wy16, NCP, myT, NCP, NV, NN1, NCP, by, nullptr, xyz, 0, 56, 0);  // y
  gemm(Wz116, NCP, gT, NCP, NZ, NB, NCP, bz1, nullptr, zBt, 1, 0, 0);   // z1
  gemm(Wz216, NP, zBt, NP, NV, NN1, NP, bz2, nullptr, xyz, 0, 112, 0);  // z2
  gemm(Ww116, NCP, gT, NCP, NZ, NB, NCP, bw1, nullptr, wBt, 1, 0, 0);   // w1
  gemm(Ww216, NP, wBt, NP, NV, NN1, NP, bw2, hw, nullptr, 2, 0, 0);     // w2

  // 4) GCN normalization
  deg_init<<<grid1(NV), 256, 0, stream>>>(deg, NV);
  deg_accum<<<grid1(NE), 256, 0, stream>>>(e_dst, deg, NE);
  deg_rsqrt<<<grid1(NV), 256, 0, stream>>>(deg, NV);
  norm_kernel<<<grid1(NE + NV), 256, 0, stream>>>(e_src, e_dst, deg, nrm, NE, NV);

  const int ROWS = NB * NV;  // 220480, multiple of 32

  // --- GCN layer 1: 168 -> 64 ---
  gemm(xyz, NXYZ, Wg1t, 192, ROWS, 64, 192, nullptr, xw, nullptr, 3, 0, 64);
  hipMemsetAsync(agg, 0, (size_t)ROWS * 64 * 4, stream);
  gcn_scatter<<<NE + NV, 256, 0, stream>>>(e_src, e_dst, nrm, xw, agg, NE, 64, 64);
  bias_lrelu<<<grid1(ROWS * 64), 256, 0, stream>>>(agg, bg1, hbuf, ROWS, 64);

  // --- GCN layer 2: 64 -> 32 ---
  gemm(hbuf, 64, Wg2t, 64, ROWS, 32, 64, nullptr, xw, nullptr, 3, 0, 32);
  hipMemsetAsync(agg, 0, (size_t)ROWS * 32 * 4, stream);
  gcn_scatter<<<NE + NV, 256, 0, stream>>>(e_src, e_dst, nrm, xw, agg, NE, 32, 32);
  bias_lrelu<<<grid1(ROWS * 32), 256, 0, stream>>>(agg, bg2, hbuf, ROWS, 32);

  // --- GCN layer 3: 32 -> 3 (N padded to 32; pad cols exactly zero) ---
  gemm(hbuf, 32, Wg3t, 32, ROWS, 32, 32, nullptr, xw, nullptr, 3, 0, 32);
  hipMemsetAsync(agg, 0, (size_t)ROWS * 32 * 4, stream);
  gcn_scatter<<<NE + NV, 256, 0, stream>>>(e_src, e_dst, nrm, xw, agg, NE, 3, 32);

  // 5) finalize: bias + sigmoid(mean(heatmap_w))
  finalize_kernel<<<grid1(ROWS), 256, 0, stream>>>(agg, bg3, hw, out, ROWS);
}
